// SimpleDenoiser_22222160789963
// MI455X (gfx1250) — compile-verified, matrix operand order checked by reading, not run on hardware
//
#include <hip/hip_runtime.h>

// ---------------------------------------------------------------------------
// SimpleDenoiser forward for MI455X (gfx1250), wave32.
// GEMMs + attention on v_wmma_f32_16x16x32_bf16; GEMM staging uses
// GLOBAL_LOAD_ASYNC_TO_LDS_B128 with ping-pong LDS (ASYNCcnt-tracked).
// ---------------------------------------------------------------------------

typedef unsigned short u16;
typedef unsigned int   u32;

typedef __bf16 v16bf __attribute__((ext_vector_type(16)));
typedef float  v8f   __attribute__((ext_vector_type(8)));

// ---------------- fp32 <-> bf16 helpers (RNE) ----------------
__device__ __forceinline__ u16 f2bf(float x) {
  union { float f; u32 u; } c; c.f = x;
  u32 u = c.u;
  u32 r = (u + 0x7fffu + ((u >> 16) & 1u)) >> 16;
  if ((u & 0x7f800000u) == 0x7f800000u) r = u >> 16;  // inf/nan passthrough
  return (u16)r;
}
__device__ __forceinline__ float bf2f(u16 h) {
  union { u32 u; float f; } c; c.u = ((u32)h) << 16;
  return c.f;
}

// ---------------- CDNA5 async global->LDS helpers ----------------
__device__ __forceinline__ void async_b128(u32 lds_addr, const void* gaddr) {
  asm volatile("global_load_async_to_lds_b128 %0, %1, off"
               :: "v"(lds_addr), "v"(gaddr) : "memory");
}
__device__ __forceinline__ void async_b128x2(u32 lds_addr, const void* gaddr) {
  asm volatile("global_load_async_to_lds_b128 %0, %1, off\n\t"
               "global_load_async_to_lds_b128 %0, %1, off offset:16"
               :: "v"(lds_addr), "v"(gaddr) : "memory");
}
__device__ __forceinline__ void wait_async0() {
  asm volatile("s_wait_asynccnt 0x0" ::: "memory");
}

// ---------------- conversion kernels ----------------
__global__ void k_cvt(const float* __restrict__ src, u16* __restrict__ dst, size_t n) {
  size_t i = (size_t)blockIdx.x * blockDim.x + threadIdx.x;
  if (i < n) dst[i] = f2bf(src[i]);
}

// (sr x sc) -> (dr x dc), zero padded
__global__ void k_cvt_pad(const float* __restrict__ src, u16* __restrict__ dst,
                          int sr, int sc, int dr, int dc) {
  size_t i = (size_t)blockIdx.x * blockDim.x + threadIdx.x;
  size_t n = (size_t)dr * dc;
  if (i >= n) return;
  int r = (int)(i / dc), c = (int)(i % dc);
  float v = (r < sr && c < sc) ? src[(size_t)r * sc + c] : 0.0f;
  dst[i] = f2bf(v);
}

// dst (sc x sr) = transpose of src (sr x sc)
__global__ void k_tcvt(const float* __restrict__ src, u16* __restrict__ dst,
                       int sr, int sc) {
  size_t i = (size_t)blockIdx.x * blockDim.x + threadIdx.x;
  size_t n = (size_t)sr * sc;
  if (i >= n) return;
  int c = (int)(i / sr), r = (int)(i % sr);     // dst row-major (sc x sr)
  dst[i] = f2bf(src[(size_t)r * sc + c]);
}

// ---------------- embedding gather: coords_bf16 (13440 x 768) ----------------
__global__ void k_embed(const float* __restrict__ svg, const float* __restrict__ wx,
                        u16* __restrict__ coords) {
  size_t tid = (size_t)blockIdx.x * blockDim.x + threadIdx.x;
  const size_t total = (size_t)13440 * 768;
  if (tid >= total) return;
  int j = (int)(tid % 768);
  int r = (int)(tid / 768);           // r = b*105 + t
  int b = r / 105, t = r % 105;
  int c6 = j >> 7, cc = j & 127;
  float sv = svg[(size_t)b * 630 + t * 6 + c6];
  float fi = (sv + 1.0f) * 0.5f * 1000.0f;
  if (fi < 0.0f) fi = 0.0f;
  if (fi > 999.0f) fi = 999.0f;
  int idx = (int)fi;
  coords[tid] = f2bf(wx[(size_t)idx * 128 + cc]);
}

// ---------------- WMMA GEMM: C[MxN] = act(A[MxK] * W[NxK]^T + bias) ----------
// Requires: M%128==0, N%64==0, K%32==0, lda/ldw even.
// Block: 256 threads = 8 waves as 4(M) x 2(N); wave tile 32x32 (2x2 WMMAs).
// Ping-pong LDS, async global->LDS staging.
#define LDSROW 20   // dwords per 32-element bf16 row (16 data + 4 pad; 16B aligned)
#define ABUF  (128 * LDSROW)
#define WBUF  (64 * LDSROW)

union AFrag { v16bf v; uint4 q[2]; };

__device__ __forceinline__ void frag_from_lds(AFrag& f, const u32* p) {
  const uint4* p4 = (const uint4*)p;
  f.q[0] = p4[0];   // K = kb .. kb+7
  f.q[1] = p4[2];   // K = 16+kb .. 16+kb+7
}

__global__ __launch_bounds__(256)
void k_gemm_bf16(const u16* __restrict__ A, int lda,
                 const u16* __restrict__ W, int ldw,
                 const float* __restrict__ bias,
                 float* __restrict__ C, int ldc,
                 u16* __restrict__ Cb, int ldcb,
                 int K, int relu) {
  __shared__ u32 ldsA[2 * ABUF];
  __shared__ u32 ldsW[2 * WBUF];

  const int tid  = threadIdx.x;
  const int lane = tid & 31;
  const int wave = tid >> 5;      // 0..7
  const int wr   = wave >> 1;     // 0..3  (wave row: 32 M-rows each)
  const int wc   = wave & 1;      // 0..1  (wave col: 32 N-cols each)

  const int m0 = blockIdx.y * 128;
  const int n0 = blockIdx.x * 64;

  // cooperative async staging: A tile 128x32 (8KB): 2 b128/thread
  //                            W tile  64x32 (4KB): 1 b128/thread
  const int arow = tid >> 1;            // 0..127
  const int abyt = (tid & 1) * 32;      // byte offset within 64B row
  const int wrow = tid >> 2;            // 0..63
  const int wbyt = (tid & 3) * 16;

  const char* gA = (const char*)(A + (size_t)(m0 + arow) * lda) + abyt;
  const char* gW = (const char*)(W + (size_t)(n0 + wrow) * ldw) + wbyt;

  const u32 lA0 = (u32)(size_t)&ldsA[0] + (u32)(arow * (LDSROW * 4) + abyt);
  const u32 lA1 = lA0 + ABUF * 4;
  const u32 lW0 = (u32)(size_t)&ldsW[0] + (u32)(wrow * (LDSROW * 4) + wbyt);
  const u32 lW1 = lW0 + WBUF * 4;

  // fragment indices (ISA 16-bit A 16x32 layout)
  const int frow = lane & 15;
  const int kb2  = (lane >> 4) << 2;    // dword offset of first K-run

  v8f acc00 = {}, acc01 = {}, acc10 = {}, acc11 = {};

  // prologue: stage chunk 0 into buffer 0
  async_b128x2(lA0, gA);
  async_b128(lW0, gW);
  gA += 64; gW += 64;
  wait_async0();
  __syncthreads();

  const int nchunks = K >> 5;
  for (int c = 0; c < nchunks; ++c) {
    const int p = c & 1;
    if (c + 1 < nchunks) {            // prefetch next chunk into other buffer
      async_b128x2(p ? lA0 : lA1, gA);
      async_b128(p ? lW0 : lW1, gW);
      gA += 64; gW += 64;
    }
    const u32* bufA = &ldsA[p * ABUF];
    const u32* bufW = &ldsW[p * WBUF];

    AFrag a0, a1, b0, b1;
    frag_from_lds(a0, bufA + (wr * 32 + frow) * LDSROW + kb2);
    frag_from_lds(a1, bufA + (wr * 32 + 16 + frow) * LDSROW + kb2);
    frag_from_lds(b0, bufW + (wc * 32 + frow) * LDSROW + kb2);
    frag_from_lds(b1, bufW + (wc * 32 + 16 + frow) * LDSROW + kb2);

    acc00 = __builtin_amdgcn_wmma_f32_16x16x32_bf16(false, a0.v, false, b0.v,
                                                    (short)0, acc00, false, false);
    acc01 = __builtin_amdgcn_wmma_f32_16x16x32_bf16(false, a0.v, false, b1.v,
                                                    (short)0, acc01, false, false);
    acc10 = __builtin_amdgcn_wmma_f32_16x16x32_bf16(false, a1.v, false, b0.v,
                                                    (short)0, acc10, false, false);
    acc11 = __builtin_amdgcn_wmma_f32_16x16x32_bf16(false, a1.v, false, b1.v,
                                                    (short)0, acc11, false, false);

    wait_async0();     // next chunk's LDS writes landed
    __syncthreads();
  }

  // epilogue: lanes 0-15 -> M=i, N=lane ; lanes 16-31 -> M=8+i, N=lane-16
  const int mbase = m0 + wr * 32 + ((lane >> 4) << 3);
  const int nbase = n0 + wc * 32 + (lane & 15);
  const float bv0 = bias ? bias[nbase] : 0.0f;
  const float bv1 = bias ? bias[nbase + 16] : 0.0f;

#pragma unroll
  for (int i = 0; i < 8; ++i) {
    float v00 = acc00[i] + bv0, v01 = acc01[i] + bv1;
    float v10 = acc10[i] + bv0, v11 = acc11[i] + bv1;
    if (relu) {
      v00 = v00 > 0.0f ? v00 : 0.0f; v01 = v01 > 0.0f ? v01 : 0.0f;
      v10 = v10 > 0.0f ? v10 : 0.0f; v11 = v11 > 0.0f ? v11 : 0.0f;
    }
    const size_t r0 = (size_t)(mbase + i);
    const size_t r1 = (size_t)(mbase + 16 + i);
    if (C) {
      C[r0 * ldc + nbase]      = v00;
      C[r0 * ldc + nbase + 16] = v01;
      C[r1 * ldc + nbase]      = v10;
      C[r1 * ldc + nbase + 16] = v11;
    }
    if (Cb) {
      Cb[r0 * ldcb + nbase]      = f2bf(v00);
      Cb[r0 * ldcb + nbase + 16] = f2bf(v01);
      Cb[r1 * ldcb + nbase]      = f2bf(v10);
      Cb[r1 * ldcb + nbase + 16] = f2bf(v11);
    }
  }
}

// ---------------- WMMA attention: one workgroup per (n, h) -----------------
// S=T=128, head_dim=16 (zero-padded to K=32 for the score WMMA).
// qkv row r = s*105 + n : q[0:128] k[128:256] v[256:384] (f32).
// Wave w owns score rows s in [16w, 16w+16): 8 score WMMAs + 4 AV WMMAs.
#define QROW 20    // dwords per Q/K LDS row (8 data + 8 zero-pad + 4 pad)
#define PROW 72    // dwords per P LDS row   (64 data + 8 pad)
#define VROW 68    // dwords per Vt LDS row  (64 data + 4 pad)

__global__ __launch_bounds__(256)
void k_attn_wmma(const float* __restrict__ qkv, u16* __restrict__ attnb) {
  __shared__ u32 sQ[128 * QROW];
  __shared__ u32 sK[128 * QROW];
  __shared__ u32 sVt[16 * VROW];
  __shared__ u32 sP[128 * PROW];

  const int n    = blockIdx.x % 105;
  const int h    = blockIdx.x / 105;
  const int tid  = threadIdx.x;
  const int lane = tid & 31;
  const int wave = tid >> 5;            // 0..7
  const int frow = lane & 15;
  const int kb2  = (lane >> 4) << 2;

  // ---- stage Q (pre-scaled by 1/sqrt(16)), K, V^T as bf16 ----
  {
    const int r  = tid >> 1;            // 0..127 : token row s (or t)
    const int hf = (tid & 1) * 8;       // d offset 0 or 8
    const float* base = qkv + ((size_t)r * 105 + n) * 384 + h * 16;
#pragma unroll
    for (int j = 0; j < 4; ++j) {
      const float q0 = base[hf + 2 * j] * 0.25f;
      const float q1 = base[hf + 2 * j + 1] * 0.25f;
      const float k0 = base[128 + hf + 2 * j];
      const float k1 = base[128 + hf + 2 * j + 1];
      sQ[r * QROW + hf / 2 + j] = (u32)f2bf(q0) | ((u32)f2bf(q1) << 16);
      sK[r * QROW + hf / 2 + j] = (u32)f2bf(k0) | ((u32)f2bf(k1) << 16);
      sQ[r * QROW + 8 + hf / 2 + j] = 0;   // zero-pad K dims 16..31
      sK[r * QROW + 8 + hf / 2 + j] = 0;
    }
    // V transpose: Vt[d][t] = V[t][d]
    u16* vt16 = (u16*)sVt;
#pragma unroll
    for (int j = 0; j < 8; ++j) {
      const int d = hf + j;
      vt16[d * (VROW * 2) + r] = f2bf(base[256 + d]);
    }
  }
  __syncthreads();

  // ---- scores: S[s][t] = (0.25 Q) . K ; wave rows = 16*wave .. +15 ----
  AFrag aq;
  frag_from_lds(aq, sQ + (wave * 16 + frow) * QROW + kb2);
  v8f sc[8];
#pragma unroll
  for (int nt = 0; nt < 8; ++nt) {
    AFrag bk;
    frag_from_lds(bk, sK + (nt * 16 + frow) * QROW + kb2);
    v8f z = {};
    sc[nt] = __builtin_amdgcn_wmma_f32_16x16x32_bf16(false, aq.v, false, bk.v,
                                                     (short)0, z, false, false);
  }

  // ---- softmax in C-layout registers: row s = 16w + ((lane>=16)?8:0) + i ---
  float rmax[8], rsum[8];
#pragma unroll
  for (int i = 0; i < 8; ++i) {
    float m = -1e30f;
#pragma unroll
    for (int nt = 0; nt < 8; ++nt) m = fmaxf(m, sc[nt][i]);
#pragma unroll
    for (int mask = 1; mask <= 8; mask <<= 1) m = fmaxf(m, __shfl_xor(m, mask, 32));
    rmax[i] = m;
  }
#pragma unroll
  for (int i = 0; i < 8; ++i) {
    float s = 0.0f;
#pragma unroll
    for (int nt = 0; nt < 8; ++nt) {
      sc[nt][i] = __expf(sc[nt][i] - rmax[i]);
      s += sc[nt][i];
    }
#pragma unroll
    for (int mask = 1; mask <= 8; mask <<= 1) s += __shfl_xor(s, mask, 32);
    rsum[i] = 1.0f / s;
  }

  // ---- write normalized P (bf16) to LDS; wave-private rows only ----
  {
    u16* p16 = (u16*)sP;
    const int mrow = wave * 16 + ((lane >> 4) << 3);
#pragma unroll
    for (int nt = 0; nt < 8; ++nt) {
#pragma unroll
      for (int i = 0; i < 8; ++i) {
        p16[(mrow + i) * (PROW * 2) + nt * 16 + frow] = f2bf(sc[nt][i] * rsum[i]);
      }
    }
  }
  // no barrier needed: each wave reads back only its own 16 P rows, and
  // same-wave LDS ops complete in order; Vt was covered by the first barrier.

  // ---- O = P @ V : K = 128 over t, N = 16 (head dim) ----
  v8f oacc = {};
#pragma unroll
  for (int kc = 0; kc < 4; ++kc) {
    AFrag ap, bv;
    frag_from_lds(ap, sP + (wave * 16 + frow) * PROW + kc * 16 + kb2);
    frag_from_lds(bv, sVt + frow * VROW + kc * 16 + kb2);
    oacc = __builtin_amdgcn_wmma_f32_16x16x32_bf16(false, ap.v, false, bv.v,
                                                   (short)0, oacc, false, false);
  }

  // ---- write O (bf16) ----
  const int mrow = wave * 16 + ((lane >> 4) << 3);
#pragma unroll
  for (int i = 0; i < 8; ++i) {
    const int s = mrow + i;
    attnb[((size_t)s * 105 + n) * 128 + h * 16 + frow] = f2bf(oacc[i]);
  }
}

// ---------------- residual + LayerNorm (wave per 128-wide row) -------------
__global__ __launch_bounds__(256)
void k_add_ln(const float* __restrict__ x, const float* __restrict__ a,
              const float* __restrict__ g, const float* __restrict__ bt,
              float* __restrict__ outf, u16* __restrict__ outb, int rows) {
  const int lane = threadIdx.x & 31;
  const int row  = blockIdx.x * (blockDim.x >> 5) + (threadIdx.x >> 5);
  if (row >= rows) return;
  const float* xr = x + (size_t)row * 128;
  const float* ar = a ? a + (size_t)row * 128 : nullptr;

  float v[4];
  float s = 0.0f;
#pragma unroll
  for (int i = 0; i < 4; ++i) {
    const int c = lane * 4 + i;
    v[i] = xr[c] + (ar ? ar[c] : 0.0f);
    s += v[i];
  }
#pragma unroll
  for (int off = 16; off; off >>= 1) s += __shfl_xor(s, off, 32);
  const float mean = s * (1.0f / 128.0f);
  float var = 0.0f;
#pragma unroll
  for (int i = 0; i < 4; ++i) { float d = v[i] - mean; var += d * d; }
#pragma unroll
  for (int off = 16; off; off >>= 1) var += __shfl_xor(var, off, 32);
  const float inv = rsqrtf(var * (1.0f / 128.0f) + 1e-5f);
#pragma unroll
  for (int i = 0; i < 4; ++i) {
    const int c = lane * 4 + i;
    const float y = (v[i] - mean) * inv * g[c] + bt[c];
    outf[(size_t)row * 128 + c] = y;
    outb[(size_t)row * 128 + c] = f2bf(y);
  }
}

// ---------------- row softmax over 1000 logits (1024-wide padded) ----------
__global__ __launch_bounds__(256)
void k_softmax(const float* __restrict__ logits, u16* __restrict__ probs) {
  __shared__ float red[16];
  const size_t r = blockIdx.x;
  const float* L = logits + r * 1024;
  const int tid = threadIdx.x, lane = tid & 31, wave = tid >> 5;

  float v[4];
  float mx = -1e30f;
#pragma unroll
  for (int i = 0; i < 4; ++i) {
    const int c = tid * 4 + i;
    v[i] = (c < 1000) ? L[c] : -1e30f;
    mx = fmaxf(mx, v[i]);
  }
#pragma unroll
  for (int off = 16; off; off >>= 1) mx = fmaxf(mx, __shfl_xor(mx, off, 32));
  if (lane == 0) red[wave] = mx;
  __syncthreads();
  float bm = red[0];
#pragma unroll
  for (int w = 1; w < 8; ++w) bm = fmaxf(bm, red[w]);

  float e[4];
  float s = 0.0f;
#pragma unroll
  for (int i = 0; i < 4; ++i) {
    const int c = tid * 4 + i;
    e[i] = (c < 1000) ? __expf(v[i] - bm) : 0.0f;
    s += e[i];
  }
#pragma unroll
  for (int off = 16; off; off >>= 1) s += __shfl_xor(s, off, 32);
  if (lane == 0) red[8 + wave] = s;
  __syncthreads();
  float bs = 0.0f;
#pragma unroll
  for (int w = 0; w < 8; ++w) bs += red[8 + w];
  const float inv = 1.0f / bs;
#pragma unroll
  for (int i = 0; i < 4; ++i)
    probs[r * 1024 + tid * 4 + i] = f2bf(e[i] * inv);
}

// ---------------- final GEMV: out[r] = dot128(nr2[r], w3) + b3 -------------
__global__ __launch_bounds__(256)
void k_final(const u16* __restrict__ A, const float* __restrict__ w,
             const float* __restrict__ b, float* __restrict__ out, int rows) {
  const int lane = threadIdx.x & 31;
  const int row  = blockIdx.x * (blockDim.x >> 5) + (threadIdx.x >> 5);
  if (row >= rows) return;
  const u16* ar = A + (size_t)row * 128;
  float s = 0.0f;
#pragma unroll
  for (int i = 0; i < 4; ++i) {
    const int c = lane * 4 + i;
    s += bf2f(ar[c]) * w[c];
  }
#pragma unroll
  for (int off = 16; off; off >>= 1) s += __shfl_xor(s, off, 32);
  if (lane == 0) out[row] = s + b[0];
}

// ===========================================================================
extern "C" void kernel_launch(void* const* d_in, const int* in_sizes, int n_in,
                              void* d_out, int out_size, void* d_ws, size_t ws_size,
                              hipStream_t stream) {
  (void)in_sizes; (void)n_in; (void)out_size; (void)ws_size;

  const float* svg            = (const float*)d_in[0];
  /* d_in[1] = timestep (unused by reference) */
  const float* w_x            = (const float*)d_in[2];
  const float* w_coords       = (const float*)d_in[3];
  const float* enc_qkv_w      = (const float*)d_in[4];
  const float* enc_qkv_b      = (const float*)d_in[5];
  const float* enc_out_w      = (const float*)d_in[6];
  const float* enc_out_b      = (const float*)d_in[7];
  const float* enc_lin1_w     = (const float*)d_in[8];
  const float* enc_lin1_b     = (const float*)d_in[9];
  const float* enc_lin2_w     = (const float*)d_in[10];
  const float* enc_lin2_b     = (const float*)d_in[11];
  const float* enc_ln1_g      = (const float*)d_in[12];
  const float* enc_ln1_bb     = (const float*)d_in[13];
  const float* enc_ln2_g      = (const float*)d_in[14];
  const float* enc_ln2_bb     = (const float*)d_in[15];
  const float* enc_norm_g     = (const float*)d_in[16];
  const float* enc_norm_b     = (const float*)d_in[17];
  const float* dec_self_qkv_w = (const float*)d_in[18];
  const float* dec_self_qkv_b = (const float*)d_in[19];
  const float* dec_self_out_w = (const float*)d_in[20];
  const float* dec_self_out_b = (const float*)d_in[21];
  const float* dec_cross_qkv_w= (const float*)d_in[22];
  const float* dec_cross_qkv_b= (const float*)d_in[23];
  const float* dec_cross_out_w= (const float*)d_in[24];
  const float* dec_cross_out_b= (const float*)d_in[25];
  const float* dec_lin1_w     = (const float*)d_in[26];
  const float* dec_lin1_b     = (const float*)d_in[27];
  const float* dec_lin2_w     = (const float*)d_in[28];
  const float* dec_lin2_b     = (const float*)d_in[29];
  const float* dec_ln1_g      = (const float*)d_in[30];
  const float* dec_ln1_bb     = (const float*)d_in[31];
  const float* dec_ln2_g      = (const float*)d_in[32];
  const float* dec_ln2_bb     = (const float*)d_in[33];
  const float* dec_ln3_g      = (const float*)d_in[34];
  const float* dec_ln3_bb     = (const float*)d_in[35];
  const float* dec_norm_g     = (const float*)d_in[36];
  const float* dec_norm_b     = (const float*)d_in[37];
  const float* mce_w1         = (const float*)d_in[38];
  const float* mce_b1         = (const float*)d_in[39];
  const float* mce_w2         = (const float*)d_in[40];
  const float* mce_b2         = (const float*)d_in[41];
  const float* mce_w3         = (const float*)d_in[42];
  const float* mce_b3         = (const float*)d_in[43];
  const float* mnr_w1         = (const float*)d_in[44];
  const float* mnr_b1         = (const float*)d_in[45];
  const float* mnr_w2         = (const float*)d_in[46];
  const float* mnr_b2         = (const float*)d_in[47];
  const float* mnr_w3         = (const float*)d_in[48];
  const float* mnr_b3         = (const float*)d_in[49];

  const int TOK = 13440;           // 128 seq * 105 batch-cols
  const int TOK2 = 80640;          // 128 * 630
  const int L = 6;

  // ---- workspace bump allocator (256B aligned) ----
  char* wp = (char*)d_ws;
  auto alloc = [&](size_t bytes) -> char* {
    char* r = wp; wp += (bytes + 255) & ~(size_t)255; return r;
  };

  // bf16 weights
  u16* wxb      = (u16*)alloc((size_t)1024 * 128 * 2);           // padded rows
  u16* wcoordsb = (u16*)alloc((size_t)128 * 768 * 2);            // transposed
  u16* eqkvb    = (u16*)alloc((size_t)L * 384 * 128 * 2);
  u16* eoutb    = (u16*)alloc((size_t)L * 128 * 128 * 2);
  u16* el1b     = (u16*)alloc((size_t)L * 2048 * 128 * 2);
  u16* el2b     = (u16*)alloc((size_t)L * 128 * 2048 * 2);
  u16* dqkvb    = (u16*)alloc((size_t)L * 384 * 128 * 2);
  u16* doutb    = (u16*)alloc((size_t)L * 128 * 128 * 2);
  u16* dcqkvb   = (u16*)alloc((size_t)L * 384 * 128 * 2);
  u16* dcoutb   = (u16*)alloc((size_t)L * 128 * 128 * 2);
  u16* dl1b     = (u16*)alloc((size_t)L * 2048 * 128 * 2);
  u16* dl2b     = (u16*)alloc((size_t)L * 128 * 2048 * 2);
  u16* m1b      = (u16*)alloc((size_t)768 * 128 * 2);
  u16* m2b      = (u16*)alloc((size_t)768 * 768 * 2);
  u16* m3b      = (u16*)alloc((size_t)768 * 768 * 2);
  u16* n1b      = (u16*)alloc((size_t)128 * 1024 * 2);           // padded cols
  u16* n2b      = (u16*)alloc((size_t)128 * 128 * 2);

  // activations
  u16*   coordsb = (u16*)alloc((size_t)TOK * 768 * 2);
  float* embf    = (float*)alloc((size_t)TOK * 128 * 4);
  u16*   embb    = (u16*)alloc((size_t)TOK * 128 * 2);
  float* xf      = (float*)alloc((size_t)TOK * 128 * 4);
  u16*   xb      = (u16*)alloc((size_t)TOK * 128 * 2);
  float* memf    = (float*)alloc((size_t)TOK * 128 * 4);
  u16*   memb    = (u16*)alloc((size_t)TOK * 128 * 2);
  float* qkvf    = (float*)alloc((size_t)TOK * 384 * 4);
  u16*   attnb   = (u16*)alloc((size_t)TOK * 128 * 2);
  float* projf   = (float*)alloc((size_t)TOK * 128 * 4);
  u16*   ffhb    = (u16*)alloc((size_t)TOK * 2048 * 2);
  u16*   t768a   = (u16*)alloc((size_t)TOK * 768 * 2);
  u16*   t768b   = (u16*)alloc((size_t)TOK * 768 * 2);
  float* logitsf = (float*)alloc((size_t)TOK2 * 1024 * 4);
  u16*   probsb  = (u16*)alloc((size_t)TOK2 * 1024 * 2);
  u16*   nr1b    = (u16*)alloc((size_t)TOK2 * 128 * 2);
  u16*   nr2b    = (u16*)alloc((size_t)TOK2 * 128 * 2);

  auto cvt = [&](const float* s, u16* d, size_t n) {
    k_cvt<<<(unsigned)((n + 255) / 256), 256, 0, stream>>>(s, d, n);
  };
  auto gemm = [&](const u16* A, int lda, const u16* W, int ldw, const float* bias,
                  float* C, int ldc, u16* Cb, int ldcb, int M, int N, int K, int relu) {
    dim3 grid((unsigned)(N / 64), (unsigned)(M / 128));
    k_gemm_bf16<<<grid, 256, 0, stream>>>(A, lda, W, ldw, bias, C, ldc, Cb, ldcb, K, relu);
  };
  auto add_ln = [&](const float* x, const float* a, const float* g, const float* bt,
                    float* of, u16* ob) {
    k_add_ln<<<TOK / 8, 256, 0, stream>>>(x, a, g, bt, of, ob, TOK);
  };

  // ---- weight conversion ----
  k_cvt_pad<<<(1024 * 128 + 255) / 256, 256, 0, stream>>>(w_x, wxb, 1000, 128, 1024, 128);
  k_tcvt<<<(768 * 128 + 255) / 256, 256, 0, stream>>>(w_coords, wcoordsb, 768, 128);
  cvt(enc_qkv_w,      eqkvb,  (size_t)L * 384 * 128);
  cvt(enc_out_w,      eoutb,  (size_t)L * 128 * 128);
  cvt(enc_lin1_w,     el1b,   (size_t)L * 2048 * 128);
  cvt(enc_lin2_w,     el2b,   (size_t)L * 128 * 2048);
  cvt(dec_self_qkv_w, dqkvb,  (size_t)L * 384 * 128);
  cvt(dec_self_out_w, doutb,  (size_t)L * 128 * 128);
  cvt(dec_cross_qkv_w,dcqkvb, (size_t)L * 384 * 128);
  cvt(dec_cross_out_w,dcoutb, (size_t)L * 128 * 128);
  cvt(dec_lin1_w,     dl1b,   (size_t)L * 2048 * 128);
  cvt(dec_lin2_w,     dl2b,   (size_t)L * 128 * 2048);
  cvt(mce_w1,         m1b,    (size_t)768 * 128);
  cvt(mce_w2,         m2b,    (size_t)768 * 768);
  cvt(mce_w3,         m3b,    (size_t)768 * 768);
  k_cvt_pad<<<(128 * 1024 + 255) / 256, 256, 0, stream>>>(mnr_w1, n1b, 128, 1000, 128, 1024);
  cvt(mnr_w2,         n2b,    (size_t)128 * 128);

  // ---- embedding: coords gather + coords @ w_coords ----
  k_embed<<<(unsigned)(((size_t)TOK * 768 + 255) / 256), 256, 0, stream>>>(svg, w_x, coordsb);
  gemm(coordsb, 768, wcoordsb, 768, nullptr, embf, 128, embb, 128, TOK, 128, 768, 0);

  hipMemcpyAsync(xf, embf, (size_t)TOK * 128 * 4, hipMemcpyDeviceToDevice, stream);
  hipMemcpyAsync(xb, embb, (size_t)TOK * 128 * 2, hipMemcpyDeviceToDevice, stream);

  // ---- encoder ----
  for (int i = 0; i < L; ++i) {
    gemm(xb, 128, eqkvb + (size_t)i * 384 * 128, 128, enc_qkv_b + i * 384,
         qkvf, 384, nullptr, 0, TOK, 384, 128, 0);
    k_attn_wmma<<<105 * 8, 256, 0, stream>>>(qkvf, attnb);
    gemm(attnb, 128, eoutb + (size_t)i * 128 * 128, 128, enc_out_b + i * 128,
         projf, 128, nullptr, 0, TOK, 128, 128, 0);
    add_ln(xf, projf, enc_ln1_g + i * 128, enc_ln1_bb + i * 128, xf, xb);
    gemm(xb, 128, el1b + (size_t)i * 2048 * 128, 128, enc_lin1_b + i * 2048,
         nullptr, 0, ffhb, 2048, TOK, 2048, 128, 1);
    gemm(ffhb, 2048, el2b + (size_t)i * 128 * 2048, 2048, enc_lin2_b + i * 128,
         projf, 128, nullptr, 0, TOK, 128, 2048, 0);
    add_ln(xf, projf, enc_ln2_g + i * 128, enc_ln2_bb + i * 128, xf, xb);
  }
  add_ln(xf, nullptr, enc_norm_g, enc_norm_b, memf, memb);

  // ---- decoder ----
  hipMemcpyAsync(xf, embf, (size_t)TOK * 128 * 4, hipMemcpyDeviceToDevice, stream);
  hipMemcpyAsync(xb, embb, (size_t)TOK * 128 * 2, hipMemcpyDeviceToDevice, stream);
  for (int i = 0; i < L; ++i) {
    // self attention
    gemm(xb, 128, dqkvb + (size_t)i * 384 * 128, 128, dec_self_qkv_b + i * 384,
         qkvf, 384, nullptr, 0, TOK, 384, 128, 0);
    k_attn_wmma<<<105 * 8, 256, 0, stream>>>(qkvf, attnb);
    gemm(attnb, 128, doutb + (size_t)i * 128 * 128, 128, dec_self_out_b + i * 128,
         projf, 128, nullptr, 0, TOK, 128, 128, 0);
    add_ln(xf, projf, dec_ln1_g + i * 128, dec_ln1_bb + i * 128, xf, xb);
    // cross attention: q from y, k/v from encoder memory
    gemm(xb, 128, dcqkvb + (size_t)i * 384 * 128, 128, dec_cross_qkv_b + i * 384,
         qkvf, 384, nullptr, 0, TOK, 128, 128, 0);
    gemm(memb, 128, dcqkvb + (size_t)i * 384 * 128 + (size_t)128 * 128, 128,
         dec_cross_qkv_b + i * 384 + 128,
         qkvf + 128, 384, nullptr, 0, TOK, 256, 128, 0);
    k_attn_wmma<<<105 * 8, 256, 0, stream>>>(qkvf, attnb);
    gemm(attnb, 128, dcoutb + (size_t)i * 128 * 128, 128, dec_cross_out_b + i * 128,
         projf, 128, nullptr, 0, TOK, 128, 128, 0);
    add_ln(xf, projf, dec_ln2_g + i * 128, dec_ln2_bb + i * 128, xf, xb);
    // FFN
    gemm(xb, 128, dl1b + (size_t)i * 2048 * 128, 128, dec_lin1_b + i * 2048,
         nullptr, 0, ffhb, 2048, TOK, 2048, 128, 1);
    gemm(ffhb, 2048, dl2b + (size_t)i * 128 * 2048, 2048, dec_lin2_b + i * 128,
         projf, 128, nullptr, 0, TOK, 128, 2048, 0);
    add_ln(xf, projf, dec_ln3_g + i * 128, dec_ln3_bb + i * 128, xf, xb);
  }
  add_ln(xf, nullptr, dec_norm_g, dec_norm_b, xf, xb);

  // ---- mce head: 128 -> 768 -> 768 -> 768 ----
  gemm(xb,    128, m1b, 128, mce_b1, nullptr, 0, t768a, 768, TOK, 768, 128, 1);
  gemm(t768a, 768, m2b, 768, mce_b2, nullptr, 0, t768b, 768, TOK, 768, 768, 1);
  gemm(t768b, 768, m3b, 768, mce_b3, nullptr, 0, t768a, 768, TOK, 768, 768, 0);

  // ---- probs = softmax(ce @ w_x^T) : ce viewed as 80640 x 128 ----
  gemm(t768a, 128, wxb, 128, nullptr, logitsf, 1024, nullptr, 0, TOK2, 1024, 128, 0);
  k_softmax<<<TOK2, 256, 0, stream>>>(logitsf, probsb);

  // ---- mnr head: 1000(pad 1024) -> 128 -> 128 -> 1 ----
  gemm(probsb, 1024, n1b, 1024, mnr_b1, nullptr, 0, nr1b, 128, TOK2, 128, 1024, 1);
  gemm(nr1b,   128,  n2b, 128,  mnr_b2, nullptr, 0, nr2b, 128, TOK2, 128, 128, 1);
  k_final<<<TOK2 / 8, 256, 0, stream>>>(nr2b, mnr_w3, mnr_b3, (float*)d_out, TOK2);
}
